// AuxInfoDCT_3418793968111
// MI455X (gfx1250) — matrix operand order repaired and checked by reading, not compiled, not run on hardware
//
#include <hip/hip_runtime.h>
#include <hip/hip_bf16.h>

// Problem constants (from reference)
#define QN 10000
#define CN 200
#define DN 64
#define HN 64
#define KN 4
#define BN 512
#define TN 200
#define MIDQ 132   // (D+C)/2
#define MIDL 132   // (H+C)/2
#define MIDD 32    // (D+1)/2

// Workspace layout:
//   xp_ws  : (T, B, 192) f32 = 78.6 MB    GRU input projection
//   qd_ws  : (B*T, 200)  f32 = 81.9 MB    qd = sigmoid(MLP(qe)); reused as qd2
//   lat_ws : (B, T, 64)  f32 = 26.2 MB    GRU hidden states
//   packed weights (f16, WMMA-B-fragment order, zero-padded): 342 KB
// total ~187.2 MB

typedef __attribute__((ext_vector_type(16))) _Float16     v16h;
typedef __attribute__((ext_vector_type(4)))  _Float16     v4h;
typedef __attribute__((ext_vector_type(8)))  float        v8f;
typedef __attribute__((ext_vector_type(4)))  float        v4f;
typedef __attribute__((ext_vector_type(4)))  unsigned int v4u;

__device__ __forceinline__ float sigf(float x) { return 1.0f / (1.0f + __expf(-x)); }

union F16Frag { v16h h; v4u u[2]; };

// A fragment from f16 LDS (row stride ldh halfs, ldh % 8 == 0): 2 ds_load_b128.
// Layout (contiguous k, matches k_pack): lane<16 row=lane holds K kc..kc+15;
// lane>=16 holds kc+16..kc+31.
__device__ __forceinline__ v16h frag_Ah(const _Float16* X, int ldh, int kc) {
  const int lane = threadIdx.x & 31;
  const int row  = lane & 15;
  const int kb   = kc + ((lane & 16) ? 16 : 0);
  const unsigned int* p = (const unsigned int*)(X + row * ldh + kb);
  F16Frag f;
  f.u[0] = *(const v4u*)(p);       // K kb..kb+7
  f.u[1] = *(const v4u*)(p + 4);   // K kb+8..kb+15
  return f.h;
}

// B fragment from pre-packed weights: 2 global_load_b128, zero VALU.
__device__ __forceinline__ v16h frag_Bp(const _Float16* Wp, int frag) {
  const int lane = threadIdx.x & 31;
  const v4u* p = (const v4u*)(Wp + (size_t)frag * 512 + lane * 16);
  F16Frag f;
  f.u[0] = p[0];
  f.u[1] = p[1];
  return f.h;
}

__device__ __forceinline__ void store4h(_Float16* dst, v4f v) {
  v4h t = {(_Float16)v[0], (_Float16)v[1], (_Float16)v[2], (_Float16)v[3]};
  *(v4h*)dst = t;
}

#define WMMA_F16(A, Bm, C) \
  __builtin_amdgcn_wmma_f32_16x16x32_f16(false, (A), false, (Bm), (short)0, (C), false, false)

// ---------------------------------------------------------------------------
// Weight packer: W (nmax x kmax, row-major f32) -> f16 fragments in per-lane
// WMMA B order: fragment (nt,kk) holds B[k][n]=W[n][k] for n in [16nt,16nt+16),
// k in [32kk,32kk+32), zero-padded. One thread per (fragment, lane) writes 16
// contiguous halfs. Requires kmax % 4 == 0 (64/132/192/320 all qualify).
// ---------------------------------------------------------------------------
__global__ void k_pack(const float* __restrict__ W, int ldw, int nmax, int kmax,
                       int kchunks, int nfrag, _Float16* __restrict__ dst) {
  int gid = blockIdx.x * blockDim.x + threadIdx.x;
  if (gid >= nfrag * 32) return;
  int frag = gid >> 5, lane = gid & 31;
  int nt = frag / kchunks, kk = frag % kchunks;
  int n  = nt * 16 + (lane & 15);
  int kb = kk * 32 + ((lane & 16) ? 16 : 0);
  int nn = (n < nmax) ? n : (nmax - 1);
  bool nok = (n < nmax);
  _Float16 vals[16];
#pragma unroll
  for (int g = 0; g < 4; ++g) {
    int kg  = kb + 4 * g;
    int kgc = (kg + 3 < kmax) ? kg : 0;
    v4f v = *(const v4f*)(W + (size_t)nn * ldw + kgc);
#pragma unroll
    for (int j = 0; j < 4; ++j)
      vals[4 * g + j] = (nok && (kg + j < kmax)) ? (_Float16)v[j] : (_Float16)0.0f;
  }
  v4h* o = (v4h*)(dst + (size_t)frag * 512 + lane * 16);
#pragma unroll
  for (int g = 0; g < 4; ++g) {
    v4h t = {vals[4 * g], vals[4 * g + 1], vals[4 * g + 2], vals[4 * g + 3]};
    o[g] = t;
  }
}

// ---------------------------------------------------------------------------
// Kernel 1: token-parallel feature pipeline. 16 tokens per workgroup, 8 waves.
// Activations live in LDS as f16 (converted once at write). Produces qd and
// xp = x@W_ih^T + b_ih in (T,B,192).
// ---------------------------------------------------------------------------
__global__ __launch_bounds__(256) void k_features(
    const float* __restrict__ E_q, const float* __restrict__ E_c,
    const float* __restrict__ E_it, const float* __restrict__ E_ut,
    const float* __restrict__ E_nh,
    const float* __restrict__ b_fuse, const float* __restrict__ b_ih,
    const float* __restrict__ qd_b1,  const float* __restrict__ qd_b2,
    const _Float16* __restrict__ qd_W1p, const _Float16* __restrict__ qd_W2p,
    const _Float16* __restrict__ W_fusep, const _Float16* __restrict__ W_ihp,
    const int* __restrict__ q_seq, const int* __restrict__ c_seq,
    const int* __restrict__ it_seq, const int* __restrict__ ut_seq,
    const int* __restrict__ nh_seq, const int* __restrict__ na_seq,
    const int* __restrict__ q2c, const int* __restrict__ q2c_mask,
    float* __restrict__ xp_ws, float* __restrict__ qd_ws)
{
  __shared__ __align__(16) _Float16 ld_x[16 * 320];    // [qe | ce | corr | E_it | fuse]
  __shared__ __align__(16) _Float16 ld_aux[16 * 192];  // [E_ut | E_nh | E_na]
  __shared__ __align__(16) _Float16 ld_h1[16 * 160];   // relu hidden (132 + zero pad)
  __shared__ __align__(16) float    ld_qd[16 * 208];   // qd (200, f32 for ce math)
  __shared__ int s_q[16], s_c[16], s_it[16], s_ut[16], s_nh[16], s_na[16];

  const int tid  = threadIdx.x;
  const int wave = tid >> 5;
  const int lane = tid & 31;
  const int base_tok = blockIdx.x * 16;   // token = b*T + t

  if (tid < 16) {
    int tk = base_tok + tid;
    s_q[tid]  = q_seq[tk];  s_c[tid]  = c_seq[tk];  s_it[tid] = it_seq[tk];
    s_ut[tid] = ut_seq[tk]; s_nh[tid] = nh_seq[tk]; s_na[tid] = na_seq[tk];
  }
  // zero K-pad of ld_h1 (cols 132..159; GEMM1 writes 0..143)
  {
    unsigned int* z = (unsigned int*)ld_h1;
#pragma unroll
    for (int i = 0; i < 5; ++i) z[tid + 256 * i] = 0u;   // 1280 uints total
  }
  __syncthreads();

  // Stage 0: vectorized gathers, f32 -> f16 at write
  {
    int i = tid >> 4, g = tid & 15;   // token, float4 index within 64 floats
    v4f qe4 = ((const v4f*)(E_q + (size_t)s_q[i] * DN))[g];
    float cf = (float)s_c[i];
    v4f c4 = {cf, cf, cf, cf};
    store4h(ld_x + i * 320 + g * 4, qe4);
    store4h(ld_x + i * 320 + 128 + g * 4, c4);
    store4h(ld_x + i * 320 + 192 + g * 4, ((const v4f*)(E_it + (size_t)s_it[i] * DN))[g]);
    store4h(ld_aux + i * 192 + g * 4,       ((const v4f*)(E_ut + (size_t)s_ut[i] * DN))[g]);
    store4h(ld_aux + i * 192 + 64 + g * 4,  ((const v4f*)(E_nh + (size_t)s_nh[i] * DN))[g]);
    store4h(ld_aux + i * 192 + 128 + g * 4, ((const v4f*)(E_nh + (size_t)s_na[i] * DN))[g]);
  }
  __syncthreads();

  // Stage 1: h1 = relu(qe @ qd_W1^T + b1)   (16x64)x(64x132)
  for (int nt = wave; nt < 9; nt += 8) {
    v8f acc = {};
#pragma unroll
    for (int kk = 0; kk < 2; ++kk)
      acc = WMMA_F16(frag_Ah(ld_x, 320, kk * 32), frag_Bp(qd_W1p, nt * 2 + kk), acc);
    int col = lane & 15, mb = (lane & 16) ? 8 : 0, n = nt * 16 + col;
    float bias = qd_b1[n < MIDQ ? n : MIDQ - 1];
    bias = (n < MIDQ) ? bias : 0.0f;
#pragma unroll
    for (int r = 0; r < 8; ++r) {
      float v = acc[r] + bias;
      ld_h1[(mb + r) * 160 + n] = (_Float16)(v > 0.0f ? v : 0.0f);
    }
  }
  __syncthreads();

  // Stage 2: qd = sigmoid(h1 @ qd_W2^T + b2)  (16x132)x(132x200)
  for (int nt = wave; nt < 13; nt += 8) {
    v8f acc = {};
#pragma unroll
    for (int kk = 0; kk < 5; ++kk)
      acc = WMMA_F16(frag_Ah(ld_h1, 160, kk * 32), frag_Bp(qd_W2p, nt * 5 + kk), acc);
    int col = lane & 15, mb = (lane & 16) ? 8 : 0, n = nt * 16 + col;
    if (n < CN) {
      float bias = qd_b2[n];
#pragma unroll
      for (int r = 0; r < 8; ++r) {
        float s = sigf(acc[r] + bias);
        ld_qd[(mb + r) * 208 + n] = s;
        qd_ws[(size_t)(base_tok + mb + r) * CN + n] = s;
      }
    }
  }
  __syncthreads();

  // Stage 3: waves 0-3 -> fuse GEMM (192->64); waves 4-7 -> concept embedding ce
  if (wave < 4) {
    v8f acc = {};
#pragma unroll
    for (int kk = 0; kk < 6; ++kk)
      acc = WMMA_F16(frag_Ah(ld_aux, 192, kk * 32), frag_Bp(W_fusep, wave * 6 + kk), acc);
    int col = lane & 15, mb = (lane & 16) ? 8 : 0, n = wave * 16 + col;
    float bias = b_fuse[n];
#pragma unroll
    for (int r = 0; r < 8; ++r)
      ld_x[(mb + r) * 320 + 256 + n] = (_Float16)(acc[r] + bias);
  } else {
    int t2 = tid - 128;            // 0..127
    int i  = t2 >> 3;              // token in tile
    int d0 = (t2 & 7) * 8;         // 8 dims each
    int q = s_q[i];
    int cc[4]; float rel[4]; float srel = 1e-6f;
#pragma unroll
    for (int k = 0; k < 4; ++k) {
      cc[k] = q2c[q * KN + k];
      float m = (float)q2c_mask[q * KN + k];
      rel[k] = ld_qd[i * 208 + cc[k]] * m;
      srel += rel[k];
    }
    float inv = 1.0f / srel;
#pragma unroll
    for (int dd = 0; dd < 8; ++dd) {
      int d = d0 + dd;
      float s = 0.0f;
#pragma unroll
      for (int k = 0; k < 4; ++k) s += rel[k] * E_c[(size_t)cc[k] * DN + d];
      ld_x[i * 320 + 64 + d] = (_Float16)(s * inv);
    }
  }
  __syncthreads();

  // Stage 4: xp = x @ W_ih^T + b_ih  (16x320)x(320x192), stored (T,B,192)
  for (int nt = wave; nt < 12; nt += 8) {
    v8f acc = {};
#pragma unroll
    for (int kk = 0; kk < 10; ++kk)
      acc = WMMA_F16(frag_Ah(ld_x, 320, kk * 32), frag_Bp(W_ihp, nt * 10 + kk), acc);
    int col = lane & 15, mb = (lane & 16) ? 8 : 0, n = nt * 16 + col;
    float bias = b_ih[n];
#pragma unroll
    for (int r = 0; r < 8; ++r) {
      int tk = base_tok + mb + r;
      int bb = tk / TN, tt = tk % TN;
      xp_ws[((size_t)tt * BN + bb) * 192 + n] = acc[r] + bias;
    }
  }
}

// ---------------------------------------------------------------------------
// Kernel 2: GRU scan. One workgroup (4 waves) owns 16 batch rows; h lives in
// LDS as f16 in the A-fragment layout (2 ds_load_b128 per fragment), W_hh
// B-fragments (pre-packed) live in registers for all 200 steps, h_old stays in
// registers, and the contiguous 12KB xp(t) tile is staged via coalesced
// float4 loads.
// ---------------------------------------------------------------------------
__global__ __launch_bounds__(128) void k_gru(
    const _Float16* __restrict__ W_hhp, const float* __restrict__ b_hh,
    const float* __restrict__ xp_ws, float* __restrict__ lat_ws)
{
  __shared__ __align__(16) _Float16 ld_h[16 * 64];
  __shared__ __align__(16) float ld_xp[16 * 192];
  const int tid = threadIdx.x, wave = tid >> 5, lane = tid & 31;
  const int b0 = blockIdx.x * 16;
  const int col = lane & 15, mb = (lane & 16) ? 8 : 0;
  const int nR = 16 * wave, nZ = 64 + 16 * wave, nNg = 128 + 16 * wave;

  // Persistent pre-packed B fragments (n-tiles: r->wave, z->4+wave, n->8+wave)
  v16h Br0 = frag_Bp(W_hhp, (wave)     * 2 + 0);
  v16h Br1 = frag_Bp(W_hhp, (wave)     * 2 + 1);
  v16h Bz0 = frag_Bp(W_hhp, (4 + wave) * 2 + 0);
  v16h Bz1 = frag_Bp(W_hhp, (4 + wave) * 2 + 1);
  v16h Bn0 = frag_Bp(W_hhp, (8 + wave) * 2 + 0);
  v16h Bn1 = frag_Bp(W_hhp, (8 + wave) * 2 + 1);
  const float bhr = b_hh[nR + col];
  const float bhz = b_hh[nZ + col];
  const float bhn = b_hh[nNg + col];

  for (int idx = tid; idx < 16 * 32; idx += 128) ((unsigned int*)ld_h)[idx] = 0u;  // h0=0
  float hprev[8];
#pragma unroll
  for (int r = 0; r < 8; ++r) hprev[r] = 0.0f;
  __syncthreads();

  for (int t = 0; t < TN; ++t) {
    // stage xp(t): 16 rows x 192 cols, contiguous 3072 floats in memory
    {
      const v4f* src = (const v4f*)(xp_ws + ((size_t)t * BN + b0) * 192);
      v4f* dst = (v4f*)ld_xp;
#pragma unroll
      for (int i = 0; i < 6; ++i) dst[tid + 128 * i] = src[tid + 128 * i];
    }
    // build A fragments from previous h (reads must precede barrier)
    v16h a0 = frag_Ah(ld_h, 64, 0);
    v16h a1 = frag_Ah(ld_h, 64, 32);
    if (t + 1 < TN)  // warm next step's xp tile in L2 (global_prefetch_b8)
      __builtin_prefetch(xp_ws + ((size_t)(t + 1) * BN + b0) * 192 + tid * 96, 0, 1);
    __syncthreads();   // ld_xp ready; all ld_h reads complete

    v8f hr = {}, hz = {}, hn = {};
    hr = WMMA_F16(a0, Br0, hr); hr = WMMA_F16(a1, Br1, hr);
    hz = WMMA_F16(a0, Bz0, hz); hz = WMMA_F16(a1, Bz1, hz);
    hn = WMMA_F16(a0, Bn0, hn); hn = WMMA_F16(a1, Bn1, hn);

    float hnew[8];
#pragma unroll
    for (int r = 0; r < 8; ++r) {
      int m = mb + r;
      float rg = sigf(ld_xp[m * 192 + nR + col] + hr[r] + bhr);
      float zg = sigf(ld_xp[m * 192 + nZ + col] + hz[r] + bhz);
      float ng = tanhf(ld_xp[m * 192 + nNg + col] + rg * (hn[r] + bhn));
      hnew[r] = (1.0f - zg) * ng + zg * hprev[r];
      ld_h[m * 64 + nR + col] = (_Float16)hnew[r];                     // next A
      lat_ws[((size_t)(b0 + m) * TN + t) * HN + nR + col] = hnew[r];   // latent
      hprev[r] = hnew[r];
    }
    __syncthreads();   // h writes visible; ld_xp reads done before next stage
  }
}

// ---------------------------------------------------------------------------
// Kernel 3: head. ua MLP via WMMA (pre-packed weights), tiny disc MLP, masked
// reduction over C=200. qd2 is read back from qd_ws (== qd[:,1:]).
// ---------------------------------------------------------------------------
__global__ __launch_bounds__(256) void k_head(
    const float* __restrict__ E_q,
    const float* __restrict__ la_b1, const float* __restrict__ la_b2,
    const _Float16* __restrict__ la_W1p, const _Float16* __restrict__ la_W2p,
    const float* __restrict__ dc_W1, const float* __restrict__ dc_b1,
    const float* __restrict__ dc_W2, const float* __restrict__ dc_b2,
    const float* __restrict__ Q_table, const int* __restrict__ q_seq,
    const float* __restrict__ lat_ws, const float* __restrict__ qd_ws,
    float* __restrict__ out)
{
  __shared__ __align__(16) _Float16 ld_lat[16 * 64];
  __shared__ __align__(16) _Float16 ld_h1[16 * 160];
  __shared__ __align__(16) float ld_ua[16 * 208];
  __shared__ float ld_disc[16];
  __shared__ int   s_b[16], s_t[16], s_q2[16];
  __shared__ float red_n[16 * 16], red_d[16 * 16];

  const int tid = threadIdx.x, wave = tid >> 5, lane = tid & 31;
  const int p0 = blockIdx.x * 16;   // pair index p = b*(T-1)+t

  if (tid < 16) {
    int p = p0 + tid;
    int b = p / (TN - 1), t = p % (TN - 1);
    s_b[tid] = b; s_t[tid] = t;
    s_q2[tid] = q_seq[b * TN + t + 1];
  }
  {
    unsigned int* z = (unsigned int*)ld_h1;
#pragma unroll
    for (int i = 0; i < 5; ++i) z[tid + 256 * i] = 0u;   // zero K-pad
  }
  __syncthreads();
  {
    int i = tid >> 4, g = tid & 15;
    store4h(ld_lat + i * 64 + g * 4,
            ((const v4f*)(lat_ws + ((size_t)s_b[i] * TN + s_t[i]) * HN))[g]);
  }
  __syncthreads();

  // ua layer 1: relu(lat @ la_W1^T + b1)
  for (int nt = wave; nt < 9; nt += 8) {
    v8f acc = {};
#pragma unroll
    for (int kk = 0; kk < 2; ++kk)
      acc = WMMA_F16(frag_Ah(ld_lat, 64, kk * 32), frag_Bp(la_W1p, nt * 2 + kk), acc);
    int col = lane & 15, mb = (lane & 16) ? 8 : 0, n = nt * 16 + col;
    float bias = la_b1[n < MIDL ? n : MIDL - 1];
    bias = (n < MIDL) ? bias : 0.0f;
#pragma unroll
    for (int r = 0; r < 8; ++r) {
      float v = acc[r] + bias;
      ld_h1[(mb + r) * 160 + n] = (_Float16)(v > 0.0f ? v : 0.0f);
    }
  }
  __syncthreads();

  // ua layer 2: sigmoid(h1 @ la_W2^T + b2)
  for (int nt = wave; nt < 13; nt += 8) {
    v8f acc = {};
#pragma unroll
    for (int kk = 0; kk < 5; ++kk)
      acc = WMMA_F16(frag_Ah(ld_h1, 160, kk * 32), frag_Bp(la_W2p, nt * 5 + kk), acc);
    int col = lane & 15, mb = (lane & 16) ? 8 : 0, n = nt * 16 + col;
    if (n < CN) {
      float bias = la_b2[n];
#pragma unroll
      for (int r = 0; r < 8; ++r) ld_ua[(mb + r) * 208 + n] = sigf(acc[r] + bias);
    }
  }

  // disc = sigmoid(mlp2(qe2)) * MAX_DISC  (tiny 64->32->1, one thread/token)
  if (tid < 16) {
    const float* qe = &E_q[(size_t)s_q2[tid] * DN];
    float acc = dc_b2[0];
    for (int j = 0; j < MIDD; ++j) {
      float s = dc_b1[j];
      for (int d = 0; d < DN; ++d) s += qe[d] * dc_W1[j * DN + d];
      acc += (s > 0.0f ? s : 0.0f) * dc_W2[j];
    }
    ld_disc[tid] = sigf(acc) * 10.0f;
  }
  __syncthreads();

  // y = disc * sum_c (ua - qd2)*Qt / (sum_c qd2*Qt + 1e-6); out = sigmoid(y)
  {
    int i = tid >> 4, s = tid & 15;
    size_t qdb = ((size_t)s_b[i] * TN + s_t[i] + 1) * CN;
    const float* Qt = &Q_table[(size_t)s_q2[i] * CN];
    float num = 0.0f, den = 0.0f;
    for (int c = s; c < CN; c += 16) {
      float qt  = Qt[c];
      float qdm = qd_ws[qdb + c] * qt;
      num += ld_ua[i * 208 + c] * qt - qdm;
      den += qdm;
    }
    red_n[i * 16 + s] = num; red_d[i * 16 + s] = den;
  }
  __syncthreads();
  if (tid < 16) {
    float num = 0.0f, den = 1e-6f;
    for (int s = 0; s < 16; ++s) { num += red_n[tid * 16 + s]; den += red_d[tid * 16 + s]; }
    out[p0 + tid] = sigf(ld_disc[tid] * num / den);
  }
}

extern "C" void kernel_launch(void* const* d_in, const int* in_sizes, int n_in,
                              void* d_out, int out_size, void* d_ws, size_t ws_size,
                              hipStream_t stream) {
  (void)in_sizes; (void)n_in; (void)out_size; (void)ws_size;
  const float* E_q    = (const float*)d_in[0];
  const float* E_c    = (const float*)d_in[1];
  const float* E_it   = (const float*)d_in[2];
  const float* E_ut   = (const float*)d_in[3];
  const float* E_nh   = (const float*)d_in[4];
  const float* W_fuse = (const float*)d_in[5];
  const float* b_fuse = (const float*)d_in[6];
  const float* W_ih   = (const float*)d_in[7];
  const float* b_ih   = (const float*)d_in[8];
  const float* W_hh   = (const float*)d_in[9];
  const float* b_hh   = (const float*)d_in[10];
  const float* qd_W1  = (const float*)d_in[11];
  const float* qd_b1  = (const float*)d_in[12];
  const float* qd_W2  = (const float*)d_in[13];
  const float* qd_b2  = (const float*)d_in[14];
  const float* la_W1  = (const float*)d_in[15];
  const float* la_b1  = (const float*)d_in[16];
  const float* la_W2  = (const float*)d_in[17];
  const float* la_b2  = (const float*)d_in[18];
  const float* dc_W1  = (const float*)d_in[19];
  const float* dc_b1  = (const float*)d_in[20];
  const float* dc_W2  = (const float*)d_in[21];
  const float* dc_b2  = (const float*)d_in[22];
  const int* q_seq    = (const int*)d_in[23];
  const int* c_seq    = (const int*)d_in[24];
  const int* it_seq   = (const int*)d_in[25];
  const int* ut_seq   = (const int*)d_in[26];
  const int* nh_seq   = (const int*)d_in[27];
  const int* na_seq   = (const int*)d_in[28];
  const int* q2c      = (const int*)d_in[29];
  const int* q2c_m    = (const int*)d_in[30];
  const float* Q_tab  = (const float*)d_in[31];

  float* xp_ws  = (float*)d_ws;                       // T*B*192
  float* qd_ws  = xp_ws + (size_t)TN * BN * 192;      // B*T*200
  float* lat_ws = qd_ws + (size_t)BN * TN * CN;       // B*T*64

  // packed weight area (f16), 16B-aligned
  _Float16* wp = (_Float16*)(lat_ws + (size_t)BN * TN * HN);
  // fragment counts: ntiles * kchunks
  const int F_qdW1 = 9 * 2, F_qdW2 = 13 * 5, F_fuse = 4 * 6, F_ih = 12 * 10;
  const int F_laW1 = 9 * 2, F_laW2 = 13 * 5, F_hh = 12 * 2;
  _Float16* qd_W1p  = wp;                    // 18 frags
  _Float16* qd_W2p  = qd_W1p  + (size_t)F_qdW1 * 512;
  _Float16* W_fusep = qd_W2p  + (size_t)F_qdW2 * 512;
  _Float16* W_ihp   = W_fusep + (size_t)F_fuse * 512;
  _Float16* la_W1p  = W_ihp   + (size_t)F_ih * 512;
  _Float16* la_W2p  = la_W1p  + (size_t)F_laW1 * 512;
  _Float16* W_hhp   = la_W2p  + (size_t)F_laW2 * 512;

  #define PACK(W, ldw, nmax, kmax, kc, nf, dst) \
    k_pack<<<((nf) * 32 + 127) / 128, 128, 0, stream>>>((W), (ldw), (nmax), (kmax), (kc), (nf), (dst))
  PACK(qd_W1, 64,  MIDQ, 64,   2,  F_qdW1, qd_W1p);
  PACK(qd_W2, MIDQ, CN,  MIDQ, 5,  F_qdW2, qd_W2p);
  PACK(W_fuse, 192, 64,  192,  6,  F_fuse, W_fusep);
  PACK(W_ih,  320, 192,  320,  10, F_ih,   W_ihp);
  PACK(la_W1, 64,  MIDL, 64,   2,  F_laW1, la_W1p);
  PACK(la_W2, MIDL, CN,  MIDL, 5,  F_laW2, la_W2p);
  PACK(W_hh,  64,  192,  64,   2,  F_hh,   W_hhp);
  #undef PACK

  k_features<<<(BN * TN) / 16, 256, 0, stream>>>(
      E_q, E_c, E_it, E_ut, E_nh, b_fuse, b_ih, qd_b1, qd_b2,
      qd_W1p, qd_W2p, W_fusep, W_ihp,
      q_seq, c_seq, it_seq, ut_seq, nh_seq, na_seq, q2c, q2c_m,
      xp_ws, qd_ws);

  k_gru<<<BN / 16, 128, 0, stream>>>(W_hhp, b_hh, xp_ws, lat_ws);

  k_head<<<(BN * (TN - 1)) / 16, 256, 0, stream>>>(
      E_q, la_b1, la_b2, la_W1p, la_W2p, dc_W1, dc_b1, dc_W2, dc_b2,
      Q_tab, q_seq, lat_ws, qd_ws, (float*)d_out);
}
